// TransformerConvStack_137438954183
// MI455X (gfx1250) — compile-verified, matrix operand order checked by reading, not run on hardware
//
#include <hip/hip_runtime.h>
#include <hip/hip_bf16.h>

typedef __attribute__((ext_vector_type(16))) _Float16 v16h;
typedef __attribute__((ext_vector_type(8)))  float    v8f;
typedef __attribute__((ext_vector_type(2)))  _Float16 v2h;
typedef __attribute__((ext_vector_type(2)))  __fp16   v2fp16;

#define NUM_NODES 10000
#define NUM_EDGES 320000
#define NODE_DIM  256
#define EDGE_DIM  64
#define HEADS     4
#define OUT_DIM   64
#define HC        256      // HEADS*OUT_DIM
#define CHUNK     20000    // 16 exact chunks of edges

union FragU { v16h v; v2h p[8]; };

__device__ __forceinline__ v2h pack_f16(float a, float b) {
  v2fp16 t = __builtin_amdgcn_cvt_pkrtz(a, b);   // v_cvt_pk_f16_f32 (RTZ)
  return __builtin_bit_cast(v2h, t);
}

// C[M x N] = A[M x K](f32, row stride lda) @ W[N x K]^T + bias, f32 out.
// f32 -> f16 conversion done in-register while building WMMA fragments.
// Block: 256 thr = 8 waves; wave w handles rows [blk*128 + w*16, +16), cols 64.
__global__ __launch_bounds__(256) void gemm_f16wmma(
    const float* __restrict__ A, int lda,
    const float* __restrict__ W,      // N x K row-major (out-features = rows)
    const float* __restrict__ bias,   // length N
    float* __restrict__ C, int ldc,
    int M, int N, int K)
{
  const int wave = threadIdx.x >> 5;
  const int lane = threadIdx.x & 31;
  const int hh   = lane >> 4;        // half-wave select
  const int l16  = lane & 15;

  const int rowTile = blockIdx.x * 128 + wave * 16;
  const int colBase = blockIdx.y * 64;

  const v8f vzero = {0.f,0.f,0.f,0.f,0.f,0.f,0.f,0.f};
  v8f acc[4];
#pragma unroll
  for (int s = 0; s < 4; ++s) acc[s] = vzero;

  int mrow = rowTile + l16;
  if (mrow >= M) mrow = M - 1;                 // clamp loads; stores guarded
  const float* arow = A + (size_t)mrow * lda;

  for (int k0 = 0; k0 < K; k0 += 32) {
    // A fragment: 16x32 f16. lane half selects K+8; VGPR i packs K pair.
    FragU af;
#pragma unroll
    for (int i = 0; i < 8; ++i) {
      const int koff = ((i < 4) ? (2*i) : (16 + 2*(i-4))) + 8*hh;
      const float2 f = *(const float2*)(arow + k0 + koff);
      af.p[i] = pack_f16(f.x, f.y);
    }
#pragma unroll
    for (int s = 0; s < 4; ++s) {
      const int n = colBase + s*16 + l16;      // N on lanes for B fragment
      const float* wrow = W + (size_t)n * K;   // B[k][n] = W[n][k]
      FragU bf;
#pragma unroll
      for (int i = 0; i < 8; ++i) {
        const int koff = ((i < 4) ? (2*i) : (16 + 2*(i-4))) + 8*hh;
        const float2 f = *(const float2*)(wrow + k0 + koff);
        bf.p[i] = pack_f16(f.x, f.y);
      }
      acc[s] = __builtin_amdgcn_wmma_f32_16x16x32_f16(
          false, af.v, false, bf.v, (short)0, acc[s], false, false);
    }
  }

  // C/D layout: VGPR r -> row r + 8*hh, col = l16
#pragma unroll
  for (int s = 0; s < 4; ++s) {
    const int n  = colBase + s*16 + l16;
    const float bv = bias[n];
#pragma unroll
    for (int r = 0; r < 8; ++r) {
      const int row = rowTile + r + 8*hh;
      if (row < M) C[(size_t)row * ldc + n] = acc[s][r] + bv;
    }
  }
}

__global__ __launch_bounds__(256) void init_attn(
    float* __restrict__ agg, float* __restrict__ mbuf, float* __restrict__ denom)
{
  const int i = blockIdx.x * blockDim.x + threadIdx.x;
  if (i < NUM_NODES * HC) agg[i] = 0.f;
  if (i < NUM_NODES * HEADS) {
    mbuf[i]  = __int_as_float(0xff800000u);  // -inf
    denom[i] = 0.f;
  }
}

// Sweep 1: logits = (q[dst] . (k[src] + e)) / sqrt(D); atomic segment max.
__global__ __launch_bounds__(256) void edge_logits_max(
    const int* __restrict__ ei, const float* __restrict__ q,
    const float* __restrict__ k, const float* __restrict__ e_chunk,
    float* __restrict__ logits, float* __restrict__ mbuf,
    int chunkStart, int chunkLen)
{
  const int idx  = blockIdx.x * blockDim.x + threadIdx.x;
  const int eL   = idx >> 2;
  const int head = idx & 3;
  if (eL >= chunkLen) return;
  const int ge = chunkStart + eL;
  const int s  = ei[ge];
  const int d  = ei[NUM_EDGES + ge];

  const float4* qp = (const float4*)(q + (size_t)d * HC + head * OUT_DIM);
  const float4* kp = (const float4*)(k + (size_t)s * HC + head * OUT_DIM);
  const float4* ep = (const float4*)(e_chunk + (size_t)eL * HC + head * OUT_DIM);

  float acc = 0.f;
#pragma unroll 4
  for (int i = 0; i < OUT_DIM / 4; ++i) {
    const float4 qv = qp[i], kv = kp[i], ev = ep[i];
    acc += qv.x*(kv.x+ev.x) + qv.y*(kv.y+ev.y) + qv.z*(kv.z+ev.z) + qv.w*(kv.w+ev.w);
  }
  const float lg = acc * 0.125f;  // 1/sqrt(64)
  logits[(size_t)ge * HEADS + head] = lg;

  float* ma = mbuf + (size_t)d * HEADS + head;
  if (lg >= 0.f) atomicMax((int*)ma, __float_as_int(lg));
  else           atomicMin((unsigned int*)ma, (unsigned int)__float_as_int(lg));
}

// Sweep 2: ex = exp(logit - m[dst]); denom += ex; agg[dst] += ex*(v[src]+e).
__global__ __launch_bounds__(256) void edge_aggregate(
    const int* __restrict__ ei, const float* __restrict__ v,
    const float* __restrict__ e_chunk, const float* __restrict__ logits,
    const float* __restrict__ mbuf, float* __restrict__ denom,
    float* __restrict__ agg, int chunkStart, int chunkLen)
{
  const int idx  = blockIdx.x * blockDim.x + threadIdx.x;
  const int eL   = idx >> 2;
  const int head = idx & 3;
  if (eL >= chunkLen) return;
  const int ge = chunkStart + eL;
  const int s  = ei[ge];
  const int d  = ei[NUM_EDGES + ge];

  const float ex = __expf(logits[(size_t)ge * HEADS + head] -
                          mbuf[(size_t)d * HEADS + head]);
  atomicAdd(denom + (size_t)d * HEADS + head, ex);

  const float4* vp = (const float4*)(v + (size_t)s * HC + head * OUT_DIM);
  const float4* ep = (const float4*)(e_chunk + (size_t)eL * HC + head * OUT_DIM);
  float* ap = agg + (size_t)d * HC + head * OUT_DIM;
#pragma unroll 4
  for (int i = 0; i < OUT_DIM / 4; ++i) {
    const float4 vv = vp[i], ev = ep[i];
    atomicAdd(ap + 4*i + 0, ex * (vv.x + ev.x));
    atomicAdd(ap + 4*i + 1, ex * (vv.y + ev.y));
    atomicAdd(ap + 4*i + 2, ex * (vv.z + ev.z));
    atomicAdd(ap + 4*i + 3, ex * (vv.w + ev.w));
  }
}

// h (already = x@Ws^T + bs) += agg / denom (per head)
__global__ __launch_bounds__(256) void finalize_nodes(
    float* __restrict__ h, const float* __restrict__ agg,
    const float* __restrict__ denom)
{
  const int i = blockIdx.x * blockDim.x + threadIdx.x;
  if (i >= NUM_NODES * HC) return;
  const int n = i >> 8;
  const int c = i & 255;
  const int head = c >> 6;
  const float dn = denom[(size_t)n * HEADS + head];
  h[i] += (dn > 0.f) ? agg[i] / dn : 0.f;
}

// cat = [h, x]  (row-major, width 512)
__global__ __launch_bounds__(256) void build_cat(
    float* __restrict__ cat, const float* __restrict__ h,
    const float* __restrict__ x)
{
  const int i = blockIdx.x * blockDim.x + threadIdx.x;
  if (i >= NUM_NODES * (HC + NODE_DIM)) return;
  const int n = i >> 9;
  const int c = i & 511;
  cat[i] = (c < HC) ? h[(size_t)n * HC + c] : x[(size_t)n * NODE_DIM + (c - HC)];
}

extern "C" void kernel_launch(void* const* d_in, const int* in_sizes, int n_in,
                              void* d_out, int out_size, void* d_ws, size_t ws_size,
                              hipStream_t stream) {
  (void)in_sizes; (void)n_in; (void)out_size; (void)ws_size;

  const float* x  = (const float*)d_in[0];
  const int*   ei = (const int*)d_in[1];
  const float* ea = (const float*)d_in[2];

  char* ws = (char*)d_ws;
  size_t off = 0;
  auto alloc = [&](size_t bytes) {
    float* p = (float*)(ws + off);
    off += (bytes + 255) & ~(size_t)255;
    return p;
  };
  float* qb     = alloc((size_t)NUM_NODES * HC * 4);
  float* kb     = alloc((size_t)NUM_NODES * HC * 4);
  float* vb     = alloc((size_t)NUM_NODES * HC * 4);
  float* hb     = alloc((size_t)NUM_NODES * HC * 4);
  float* agg    = alloc((size_t)NUM_NODES * HC * 4);
  float* cat    = alloc((size_t)NUM_NODES * (HC + NODE_DIM) * 4);
  float* logits = alloc((size_t)NUM_EDGES * HEADS * 4);
  float* mbuf   = alloc((size_t)NUM_NODES * HEADS * 4);
  float* denom  = alloc((size_t)NUM_NODES * HEADS * 4);
  float* echunk = alloc((size_t)CHUNK * HC * 4);

  const dim3 blk(256);
  const dim3 gN((NUM_NODES + 127) / 128, HC / 64);     // node GEMMs -> 256 out
  const dim3 gE((CHUNK + 127) / 128, HC / 64);         // edge GEMM chunks
  const int  nEdgeThr = (CHUNK * HEADS + 255) / 256;

  for (int l = 0; l < 3; ++l) {
    const float* Ain = (l == 0) ? x : cat;
    const int    K   = (l == 0) ? NODE_DIM : (NODE_DIM + HC);
    const int    b   = 3 + 10 * l;
    const float* Wq = (const float*)d_in[b + 0];
    const float* bq = (const float*)d_in[b + 1];
    const float* Wk = (const float*)d_in[b + 2];
    const float* bk = (const float*)d_in[b + 3];
    const float* Wv = (const float*)d_in[b + 4];
    const float* bv = (const float*)d_in[b + 5];
    const float* We = (const float*)d_in[b + 6];
    const float* be = (const float*)d_in[b + 7];
    const float* Wsk = (const float*)d_in[b + 8];
    const float* bsk = (const float*)d_in[b + 9];

    gemm_f16wmma<<<gN, blk, 0, stream>>>(Ain, K, Wq, bq, qb, HC, NUM_NODES, HC, K);
    gemm_f16wmma<<<gN, blk, 0, stream>>>(Ain, K, Wk, bk, kb, HC, NUM_NODES, HC, K);
    gemm_f16wmma<<<gN, blk, 0, stream>>>(Ain, K, Wv, bv, vb, HC, NUM_NODES, HC, K);
    gemm_f16wmma<<<gN, blk, 0, stream>>>(Ain, K, Wsk, bsk, hb, HC, NUM_NODES, HC, K);

    init_attn<<<(NUM_NODES * HC + 255) / 256, blk, 0, stream>>>(agg, mbuf, denom);

    // Pass 1: per-edge logits + segment max (e recomputed per chunk via WMMA)
    for (int c0 = 0; c0 < NUM_EDGES; c0 += CHUNK) {
      gemm_f16wmma<<<gE, blk, 0, stream>>>(ea + (size_t)c0 * EDGE_DIM, EDGE_DIM,
                                           We, be, echunk, HC, CHUNK, HC, EDGE_DIM);
      edge_logits_max<<<nEdgeThr, blk, 0, stream>>>(ei, qb, kb, echunk, logits,
                                                    mbuf, c0, CHUNK);
    }
    // Pass 2: exp / denom / unnormalized aggregation
    for (int c0 = 0; c0 < NUM_EDGES; c0 += CHUNK) {
      gemm_f16wmma<<<gE, blk, 0, stream>>>(ea + (size_t)c0 * EDGE_DIM, EDGE_DIM,
                                           We, be, echunk, HC, CHUNK, HC, EDGE_DIM);
      edge_aggregate<<<nEdgeThr, blk, 0, stream>>>(ei, vb, echunk, logits, mbuf,
                                                   denom, agg, c0, CHUNK);
    }

    finalize_nodes<<<(NUM_NODES * HC + 255) / 256, blk, 0, stream>>>(hb, agg, denom);

    if (l < 2)
      build_cat<<<(NUM_NODES * (HC + NODE_DIM) + 255) / 256, blk, 0, stream>>>(cat, hb, x);
  }

  // Final linear: out = h @ Wl^T + bl  (N = 64)
  const float* Wl = (const float*)d_in[33];
  const float* bl = (const float*)d_in[34];
  const dim3 gF((NUM_NODES + 127) / 128, 1);
  gemm_f16wmma<<<gF, blk, 0, stream>>>(hb, HC, Wl, bl, (float*)d_out, OUT_DIM,
                                       NUM_NODES, OUT_DIM, HC);
}
